// SimpleGCN_68281390072316
// MI455X (gfx1250) — matrix operand order repaired and checked by reading, not compiled, not run on hardware
//
#include <hip/hip_runtime.h>
#include <hip/hip_bf16.h>
#include <stdint.h>

// ---------------------------------------------------------------------------
// SimpleGCN on MI455X (gfx1250, wave32).
//  - Edge scatter dominates (~1.4 GB/pass, L2-resident @192MB): float4 row
//    reads + hardware f32 atomics, feature-contiguous thread mapping.
//  - Dense h@W via V_WMMA_F32_16X16X32_BF16, one wave per 16x16 tile,
//    layer shapes templated so the K loop fully unrolls and addressing is
//    immediate-offset. W pre-packed to this code's B-operand bf16 layout (contiguous k).
//  Operand layouts used here (A/B k order is contiguous, not the ISA table order):
//    A (16x32 bf16): lane m=lane&15; element j -> K=16*hi+j
//    B (32x16 bf16): lane n=lane&15; element j -> K=16*hi+j
//    C/D (16x16 f32): vgpr r -> M=r+8*hi, N=lane&15
// ---------------------------------------------------------------------------

#define THREADS 256

typedef __attribute__((ext_vector_type(16))) __bf16 v16bf;
typedef __attribute__((ext_vector_type(8)))  float  v8f;

__global__ void gcn_init_deg(float* __restrict__ deg, int n) {
  int i = blockIdx.x * blockDim.x + threadIdx.x;
  if (i < n) deg[i] = 1.0f;                   // self-loop
}

__global__ void gcn_count_deg(const int* __restrict__ dst,
                              float* __restrict__ deg, int e) {
  int i = blockIdx.x * blockDim.x + threadIdx.x;
  if (i < e) unsafeAtomicAdd(&deg[dst[i]], 1.0f);
}

__global__ void gcn_norms(const float* __restrict__ deg,
                          float* __restrict__ dis,
                          float* __restrict__ sn, int n) {
  int i = blockIdx.x * blockDim.x + threadIdx.x;
  if (i < n) {
    float d = deg[i];
    dis[i] = rsqrtf(d);
    sn[i]  = 1.0f / d;
  }
}

__global__ void gcn_edge_norm(const int* __restrict__ src,
                              const int* __restrict__ dst,
                              const float* __restrict__ dis,
                              float* __restrict__ en, int e) {
  int i = blockIdx.x * blockDim.x + threadIdx.x;
  if (i < e) en[i] = dis[src[i]] * dis[dst[i]];
}

// Pre-pack W[K x DOUT] (f32 row-major) into per-(k-chunk, n-tile) B fragments:
// fragment (kc,nc) = 32 lanes x 16 bf16 contiguous; lane's 16 values are its
// column n = nc*16 + (lane&15) at K = kc*32 + 16*(lane>>4) + j.
template <int K, int DOUT>
__global__ void gcn_pack_w(const float* __restrict__ W, __bf16* __restrict__ Wp) {
  constexpr int TILES = (K / 32) * (DOUT / 16);
  int t = blockIdx.x * blockDim.x + threadIdx.x;
  if (t >= TILES * 32) return;
  int tile = t >> 5, lane = t & 31;
  int kc = tile / (DOUT / 16), nc = tile % (DOUT / 16);
  int hi = lane >> 4, lm = lane & 15;
  const float* bp = W + (size_t)(kc * 32 + hi * 16) * DOUT + nc * 16 + lm;
  __bf16* op = Wp + ((size_t)tile * 32 + lane) * 16;
#pragma unroll
  for (int j = 0; j < 16; ++j) op[j] = (__bf16)bp[(size_t)j * DOUT];
}

// T[nrows x DOUT] = (RELU ? max(A,0) : A)[nrows x K] @ W, bf16 WMMA, f32 acc.
template <int K, int DOUT, bool RELU>
__global__ void gcn_gemm_wmma(const float* __restrict__ A,
                              const __bf16* __restrict__ Wp,
                              float* __restrict__ T, int nrows) {
  constexpr int TN = DOUT >> 4;
  const int lane = threadIdx.x & 31;
  const int wave = (int)((blockIdx.x * blockDim.x + threadIdx.x) >> 5);
  const int nTiles = (nrows >> 4) * TN;
  if (wave >= nTiles) return;                 // wave-uniform: EXEC stays all-1s
  const int m0 = (wave / TN) << 4;
  const int nc = wave % TN;
  const int hi = lane >> 4;
  const int lm = lane & 15;

  const float*  arow  = A + (size_t)(m0 + lm) * K + (hi << 4);
  const __bf16* bbase = Wp + ((size_t)nc * 32 + lane) * 16;
  v8f acc = {};

#pragma unroll
  for (int kc = 0; kc < K / 32; ++kc) {
    const int k0 = kc * 32;
    float4 a0 = *(const float4*)(arow + k0);
    float4 a1 = *(const float4*)(arow + k0 + 4);
    float4 a2 = *(const float4*)(arow + k0 + 8);
    float4 a3 = *(const float4*)(arow + k0 + 12);
    if (RELU) {
      a0.x = fmaxf(a0.x, 0.f); a0.y = fmaxf(a0.y, 0.f);
      a0.z = fmaxf(a0.z, 0.f); a0.w = fmaxf(a0.w, 0.f);
      a1.x = fmaxf(a1.x, 0.f); a1.y = fmaxf(a1.y, 0.f);
      a1.z = fmaxf(a1.z, 0.f); a1.w = fmaxf(a1.w, 0.f);
      a2.x = fmaxf(a2.x, 0.f); a2.y = fmaxf(a2.y, 0.f);
      a2.z = fmaxf(a2.z, 0.f); a2.w = fmaxf(a2.w, 0.f);
      a3.x = fmaxf(a3.x, 0.f); a3.y = fmaxf(a3.y, 0.f);
      a3.z = fmaxf(a3.z, 0.f); a3.w = fmaxf(a3.w, 0.f);
    }
    v16bf av;
    av[0]  = (__bf16)a0.x; av[1]  = (__bf16)a0.y;
    av[2]  = (__bf16)a0.z; av[3]  = (__bf16)a0.w;
    av[4]  = (__bf16)a1.x; av[5]  = (__bf16)a1.y;
    av[6]  = (__bf16)a1.z; av[7]  = (__bf16)a1.w;
    av[8]  = (__bf16)a2.x; av[9]  = (__bf16)a2.y;
    av[10] = (__bf16)a2.z; av[11] = (__bf16)a2.w;
    av[12] = (__bf16)a3.x; av[13] = (__bf16)a3.y;
    av[14] = (__bf16)a3.z; av[15] = (__bf16)a3.w;

    v16bf bv = *(const v16bf*)(bbase + (size_t)kc * (TN * 32 * 16));

    acc = __builtin_amdgcn_wmma_f32_16x16x32_bf16(
        /*neg_a=*/false, av, /*neg_b=*/false, bv,
        /*c_mod=*/(short)0, acc, /*reuse_a=*/false, /*reuse_b=*/false);
  }

  float* orow = T + (size_t)(m0 + (hi << 3)) * DOUT + (nc << 4) + lm;
#pragma unroll
  for (int r = 0; r < 8; ++r) orow[(size_t)r * DOUT] = acc[r];
}

// out[i..i+3] = T[i..i+3]*self_norm[row] + bias[col..col+3]
__global__ void gcn_agg_init4(const float* __restrict__ T,
                              const float* __restrict__ sn,
                              const float* __restrict__ bias,
                              float* __restrict__ out, int total4, int shift) {
  int i = blockIdx.x * blockDim.x + threadIdx.x;
  if (i >= total4) return;
  int base = i << 2;
  int row = base >> shift;
  int col = base & ((1 << shift) - 1);
  float4 t = *(const float4*)(T + base);
  float4 b = *(const float4*)(bias + col);
  float s = sn[row];
  float4 o;
  o.x = t.x * s + b.x; o.y = t.y * s + b.y;
  o.z = t.z * s + b.z; o.w = t.w * s + b.w;
  *(float4*)(out + base) = o;
}

// out[dst[e]][d..d+3] += T[src[e]][d..d+3] * enorm[e]   (hardware f32 atomics)
__global__ void gcn_agg_edges4(const int* __restrict__ src,
                               const int* __restrict__ dst,
                               const float* __restrict__ en,
                               const float* __restrict__ T,
                               float* __restrict__ out, int total4, int qshift) {
  int i = blockIdx.x * blockDim.x + threadIdx.x;
  if (i >= total4) return;
  int e = i >> qshift;
  int q = i & ((1 << qshift) - 1);
  size_t soff = (((size_t)src[e]) << (qshift + 2)) + (q << 2);
  size_t doff = (((size_t)dst[e]) << (qshift + 2)) + (q << 2);
  float w = en[e];
  float4 t = *(const float4*)(T + soff);
  float* o = out + doff;
  unsafeAtomicAdd(o + 0, t.x * w);
  unsafeAtomicAdd(o + 1, t.y * w);
  unsafeAtomicAdd(o + 2, t.z * w);
  unsafeAtomicAdd(o + 3, t.w * w);
}

static inline unsigned nblocks(long long t) {
  return (unsigned)((t + THREADS - 1) / THREADS);
}

extern "C" void kernel_launch(void* const* d_in, const int* in_sizes, int n_in,
                              void* d_out, int out_size, void* d_ws, size_t ws_size,
                              hipStream_t stream) {
  const float* x  = (const float*)d_in[0];
  const int*   ei = (const int*)d_in[1];     // [2,E] flat: row0=src, row1=dst
  const float* W1 = (const float*)d_in[2];
  const float* b1 = (const float*)d_in[3];
  const float* W2 = (const float*)d_in[4];
  const float* b2 = (const float*)d_in[5];
  const float* W3 = (const float*)d_in[6];
  const float* b3 = (const float*)d_in[7];
  float* out = (float*)d_out;

  constexpr int D = 256, D1 = 64, D2 = 32, D3 = 16;
  const int N = in_sizes[0] / D;             // 100000
  const int E = in_sizes[1] / 2;             // 1600000

  const int* src = ei;
  const int* dst = ei + E;

  float* ws   = (float*)d_ws;
  float* deg  = ws;                          // N
  float* dis  = deg + N;                     // N
  float* sn   = dis + N;                     // N
  float* en   = sn + N;                      // E
  float* bufT = en + E;                      // N*D1 (GEMM out, per layer)
  float* bufH = bufT + (size_t)N * D1;       // N*D1 (aggregated h, per layer)
  __bf16* wp1 = (__bf16*)(bufH + (size_t)N * D1);   // 32B-aligned
  __bf16* wp2 = wp1 + (size_t)D  * D1;
  __bf16* wp3 = wp2 + (size_t)D1 * D2;

  // --- degree / normalization / weight packing ---
  gcn_init_deg <<<nblocks(N), THREADS, 0, stream>>>(deg, N);
  gcn_count_deg<<<nblocks(E), THREADS, 0, stream>>>(dst, deg, E);
  gcn_norms    <<<nblocks(N), THREADS, 0, stream>>>(deg, dis, sn, N);
  gcn_edge_norm<<<nblocks(E), THREADS, 0, stream>>>(src, dst, dis, en, E);
  gcn_pack_w<D,  D1><<<nblocks((D  / 32) * (D1 / 16) * 32), THREADS, 0, stream>>>(W1, wp1);
  gcn_pack_w<D1, D2><<<nblocks((D1 / 32) * (D2 / 16) * 32), THREADS, 0, stream>>>(W2, wp2);
  gcn_pack_w<D2, D3><<<nblocks((D2 / 32) * (D3 / 16) * 32), THREADS, 0, stream>>>(W3, wp3);

  // --- layer 1: 256 -> 64 ---
  gcn_gemm_wmma<D, D1, false>
      <<<nblocks((long long)(N >> 4) * (D1 >> 4) * 32), THREADS, 0, stream>>>(x, wp1, bufT, N);
  gcn_agg_init4 <<<nblocks((long long)N * D1 / 4), THREADS, 0, stream>>>(bufT, sn, b1, bufH, N * (D1 / 4), 6);
  gcn_agg_edges4<<<nblocks((long long)E * D1 / 4), THREADS, 0, stream>>>(src, dst, en, bufT, bufH, E * (D1 / 4), 4);

  // --- layer 2: 64 -> 32 (ReLU fused into A-load) ---
  gcn_gemm_wmma<D1, D2, true>
      <<<nblocks((long long)(N >> 4) * (D2 >> 4) * 32), THREADS, 0, stream>>>(bufH, wp2, bufT, N);
  gcn_agg_init4 <<<nblocks((long long)N * D2 / 4), THREADS, 0, stream>>>(bufT, sn, b2, bufH, N * (D2 / 4), 5);
  gcn_agg_edges4<<<nblocks((long long)E * D2 / 4), THREADS, 0, stream>>>(src, dst, en, bufT, bufH, E * (D2 / 4), 3);

  // --- layer 3: 32 -> 16 (ReLU fused), output straight to d_out ---
  gcn_gemm_wmma<D2, D3, true>
      <<<nblocks((long long)(N >> 4) * (D3 >> 4) * 32), THREADS, 0, stream>>>(bufH, wp3, bufT, N);
  gcn_agg_init4 <<<nblocks((long long)N * D3 / 4), THREADS, 0, stream>>>(bufT, sn, b3, out, N * (D3 / 4), 4);
  gcn_agg_edges4<<<nblocks((long long)E * D3 / 4), THREADS, 0, stream>>>(src, dst, en, bufT, out, E * (D3 / 4), 2);
}
